// GSensorResponse_687194768379
// MI455X (gfx1250) — matrix operand order repaired and checked by reading, not compiled, not run on hardware
//
#include <hip/hip_runtime.h>
#include <hip/hip_bf16.h>

// ---------------------------------------------------------------------------
// GSensorResponse on MI455X (gfx1250, wave32, WMMA + async global->LDS)
//
// out[p,t] = sum_k  sr[k,p] * ev[k,t]     (M=2304, N=576(pad of 550), K=24000)
// sr[k,p]  = exp(-0.5*r2/var)*(INV_2PI/var)*resp[k]   (generated in-kernel)
// ev[k,t]  = exp(-(t-z)^2/(2*bs))*GNORM/sqrt(bs)      (precomputed, f16, t-major)
// resp[k]  = exp(MLP(x)) * mask                        (precomputed, f32)
// ---------------------------------------------------------------------------

typedef __attribute__((ext_vector_type(16))) _Float16 v16h;
typedef __attribute__((ext_vector_type(8)))  float    v8f;

#define NE      24000      // B*N events (K dimension)
#define SS      2304       // 48*48 sensors (M dimension)
#define TT      550        // ticks
#define TPAD    576        // padded ticks (3 * 192)
#define HH      128        // MLP hidden
#define KCHUNKS 750        // NE / 32
#define NSLICE  8          // k-split
#define CPS     94         // ceil(KCHUNKS / NSLICE)

#define INV_2PI   (1.0f / (2.0f * 3.14159f))
#define GNORM     0.3989422804f

// workspace byte offsets (all 256B aligned)
#define WS_RESP 0u                 // 24000 f32            =    96,000 B
#define WS_EVT  96256u             // 576*24000 f16        = 27,648,000 B
#define WS_PART 27744256u          // 8*2304*576 f32       = 42,467,328 B

union H2   { _Float16 h[2]; unsigned u; };
union Frag { unsigned u[8]; v16h h; };

// ---------------------------------------------------------------------------
// Kernel 1: per-event MLP -> resp[e].  One wave32 per event, 8 events/block.
// ---------------------------------------------------------------------------
__global__ void __launch_bounds__(256)
mlp_resp_kernel(const float* __restrict__ xin, const float* __restrict__ mask,
                const float* __restrict__ W1, const float* __restrict__ b1,
                const float* __restrict__ W2, const float* __restrict__ b2,
                const float* __restrict__ W3, const float* __restrict__ b3,
                float* __restrict__ resp) {
  __shared__ float sh_h[8][HH];
  const int wave = threadIdx.x >> 5;
  const int lane = threadIdx.x & 31;
  const int e    = blockIdx.x * 8 + wave;

  const float x = xin[2 * e], y = xin[2 * e + 1];
  const int j0 = lane * 4;

  // layer 1: 2 -> 128, each lane owns 4 consecutive hidden units
  const float4 w1a = *(const float4*)(W1 + j0);        // W1[0][j]
  const float4 w1b = *(const float4*)(W1 + HH + j0);   // W1[1][j]
  const float4 bb1 = *(const float4*)(b1 + j0);
  sh_h[wave][j0 + 0] = fmaxf(0.f, x * w1a.x + y * w1b.x + bb1.x);
  sh_h[wave][j0 + 1] = fmaxf(0.f, x * w1a.y + y * w1b.y + bb1.y);
  sh_h[wave][j0 + 2] = fmaxf(0.f, x * w1a.z + y * w1b.z + bb1.z);
  sh_h[wave][j0 + 3] = fmaxf(0.f, x * w1a.w + y * w1b.w + bb1.w);
  __syncthreads();

  // layer 2: 128 -> 128
  float4 acc = *(const float4*)(b2 + j0);
  for (int k = 0; k < HH; ++k) {
    const float hv = sh_h[wave][k];
    const float4 w = *(const float4*)(W2 + k * HH + j0);
    acc.x += hv * w.x; acc.y += hv * w.y; acc.z += hv * w.z; acc.w += hv * w.w;
  }
  acc.x = fmaxf(acc.x, 0.f); acc.y = fmaxf(acc.y, 0.f);
  acc.z = fmaxf(acc.z, 0.f); acc.w = fmaxf(acc.w, 0.f);

  // layer 3: 128 -> 1, shuffle reduce across the wave
  const float4 w3 = *(const float4*)(W3 + j0);
  float s = acc.x * w3.x + acc.y * w3.y + acc.z * w3.z + acc.w * w3.w;
  #pragma unroll
  for (int off = 16; off > 0; off >>= 1) s += __shfl_down(s, off, 32);
  if (lane == 0) resp[e] = expf(s + b3[0]) * mask[e];
}

// ---------------------------------------------------------------------------
// Kernel 2: temporal gaussian, stored transposed + padded: evT[t][k], f16
// ---------------------------------------------------------------------------
__global__ void __launch_bounds__(256)
ev_kernel(const float* __restrict__ z, const float* __restrict__ nbs,
          _Float16* __restrict__ evT) {
  const int idx = blockIdx.x * 256 + threadIdx.x;     // t*NE + k
  const int t = idx / NE;
  const int k = idx - t * NE;
  const float bs = nbs[0] * nbs[0];
  float val = 0.f;
  if (t < TT) {
    const float dz = (float)t - z[k];
    val = __expf(-dz * dz / (2.f * bs)) * (GNORM * rsqrtf(bs));
  }
  evT[idx] = (_Float16)val;
}

// ---------------------------------------------------------------------------
// Kernel 3: fused sr-generation + WMMA GEMM with deterministic k-split.
// Block = 256 threads (8 waves). Tile = 64p x 192t. Wave -> 16p x 96t
// (6 C tiles of 16x16, f32 accum). K chunk = 32 events per step.
// B tile staged with GLOBAL_LOAD_ASYNC_TO_LDS_B128 (ASYNCcnt), double-buffered
// so the next chunk's transfer overlaps this chunk's exp + WMMA work.
// sched_barrier(0) fences keep the fragment ds_loads grouped ahead of the
// WMMA burst (distinct registers, one s_wait_dscnt, 6 back-to-back WMMAs).
// ---------------------------------------------------------------------------
__global__ void __launch_bounds__(256)
gemm_kernel(const float* __restrict__ xin,  const float* __restrict__ sloc,
            const float* __restrict__ elsp, const float* __restrict__ resp,
            const unsigned* __restrict__ evT_u32,   // evT as packed f16x2
            float* __restrict__ part) {
  __shared__ __align__(16) unsigned Bs[2][192 * 16]; // [buf][t_local][kpair] 24 KB
  __shared__ __align__(16) unsigned As[2][64 * 16];  // [buf][p_local][kpair]  8 KB
  __shared__ float evx[2][32], evy[2][32], evr[2][32];

  const int tid = threadIdx.x;
  const int p0  = blockIdx.x * 64;
  const int t0  = blockIdx.y * 192;
  const int ks  = blockIdx.z;
  const int c0  = ks * CPS;
  const int c1  = (c0 + CPS < KCHUNKS) ? (c0 + CPS) : KCHUNKS;

  const float var   = elsp[0] * elsp[0];
  const float inv2v = -0.5f / var;
  const float coef  = INV_2PI / var;

  const int wave  = tid >> 5,  lane  = tid & 31;
  const int psub  = wave & 3,  thalf = wave >> 2;   // wave -> (16p, 96t) region
  const int m     = lane & 15, half  = lane >> 4;   // WMMA lane decomposition
  const int pairK = tid & 15,  prow  = tid >> 4;    // A-gen mapping

  // Per-thread async-copy addressing: thread covers 3 uint4 slots of the
  // 192x16-u32 B tile.  q = tid + j*256, row = q>>2, col4 = q&3.
  const int q0  = tid;
  const int tr0 = q0 >> 2, c40 = q0 & 3;
  const unsigned ldsB0 = (unsigned)(uintptr_t)(&Bs[0][0]);
  const unsigned ldsB1 = (unsigned)(uintptr_t)(&Bs[1][0]);

  #define ISSUE_ASYNC_B(chunk, buf)                                           \
    {                                                                         \
      const unsigned* bsrc_ =                                                 \
          evT_u32 + (size_t)t0 * (NE / 2) + (chunk) * 16;                     \
      const unsigned ldsb_ = (buf) ? ldsB1 : ldsB0;                           \
      _Pragma("unroll")                                                       \
      for (int j_ = 0; j_ < 3; ++j_) {                                        \
        const int q_ = q0 + j_ * 256;                                         \
        const int tr_ = tr0 + j_ * 64;  /* (q0+j*256)>>2 */                   \
        const unsigned* g_ = bsrc_ + tr_ * (NE / 2) + c40 * 4;                \
        const unsigned l_ = ldsb_ + (unsigned)(q_ * 16);                      \
        asm volatile("global_load_async_to_lds_b128 %0, %1, off"             \
                     :: "v"(l_), "v"(g_) : "memory");                         \
      }                                                                       \
    }

  v8f c[6] = {};

  // ---- prologue: kick off chunk c0's B transfer + event staging ----
  ISSUE_ASYNC_B(c0, 0)
  if (tid < 32) {
    const int e = c0 * 32 + tid;
    evx[0][tid] = xin[2 * e];
    evy[0][tid] = xin[2 * e + 1];
    evr[0][tid] = resp[e];
  }

  for (int ch = c0; ch < c1; ++ch) {
    const int buf = (ch - c0) & 1;
    const bool have_next = (ch + 1 < c1);

    __syncthreads();   // events(ch) visible; prev iter's frag reads complete

    // --- prefetch next chunk into the other buffer (overlaps exp + WMMA) ---
    if (have_next) {
      ISSUE_ASYNC_B(ch + 1, buf ^ 1)
      if (tid < 32) {
        const int e = (ch + 1) * 32 + tid;
        evx[buf ^ 1][tid] = xin[2 * e];
        evy[buf ^ 1][tid] = xin[2 * e + 1];
        evr[buf ^ 1][tid] = resp[e];
      }
    }

    // --- generate A tile: sr for 64 p's x 32 k's, packed f16x2 along k ---
    {
      const float ax0 = evx[buf][2 * pairK],     ay0 = evy[buf][2 * pairK];
      const float ar0 = evr[buf][2 * pairK];
      const float ax1 = evx[buf][2 * pairK + 1], ay1 = evy[buf][2 * pairK + 1];
      const float ar1 = evr[buf][2 * pairK + 1];
      #pragma unroll
      for (int i = 0; i < 4; ++i) {
        const int pl = prow + i * 16;
        const int p  = p0 + pl;
        const float sx = sloc[2 * p], sy = sloc[2 * p + 1];
        const float dx0 = ax0 - sx, dy0 = ay0 - sy;
        const float dx1 = ax1 - sx, dy1 = ay1 - sy;
        const float v0 = __expf((dx0 * dx0 + dy0 * dy0) * inv2v) * coef * ar0;
        const float v1 = __expf((dx1 * dx1 + dy1 * dy1) * inv2v) * coef * ar1;
        H2 pk; pk.h[0] = (_Float16)v0; pk.h[1] = (_Float16)v1;
        As[buf][pl * 16 + pairK] = pk.u;
      }
    }

    // --- retire this chunk's 3 async transfers (next chunk's stay in flight) ---
    if (have_next) asm volatile("s_wait_asynccnt 0x3" ::: "memory");
    else           asm volatile("s_wait_asynccnt 0x0" ::: "memory");
    __syncthreads();   // A + B tiles visible to all waves

    // --- fragment loads grouped ahead of the WMMA burst ---
    __builtin_amdgcn_sched_barrier(0);

    Frag af;
    const int arow = (psub * 16 + m) * 16;
    #pragma unroll
    for (int v = 0; v < 8; ++v)   // A 16x32: K = half*16 + 2v
      af.u[v] = As[buf][arow + half * 8 + v];

    Frag bf[6];
    #pragma unroll
    for (int s = 0; s < 6; ++s) {
      const int trow = (thalf * 96 + s * 16 + m) * 16 + half * 8;
      #pragma unroll
      for (int v = 0; v < 8; ++v) // B 32x16: K = half*16 + 2v, col = lane%16
        bf[s].u[v] = Bs[buf][trow + v];
    }

    __builtin_amdgcn_sched_barrier(0);   // no ds_load may sink past here

    #pragma unroll
    for (int s = 0; s < 6; ++s)
      c[s] = __builtin_amdgcn_wmma_f32_16x16x32_f16(
                 false, af.h, false, bf[s].h, (short)0, c[s], false, false);

    __builtin_amdgcn_sched_barrier(0);   // keep the burst intact
  }

  // --- write this k-slice's partial tile (C/D layout: VGPR r -> M=half*8+r) ---
  #pragma unroll
  for (int s = 0; s < 6; ++s) {
    const int t = t0 + thalf * 96 + s * 16 + m;
    #pragma unroll
    for (int r = 0; r < 8; ++r) {
      const int p = p0 + psub * 16 + half * 8 + r;
      part[((size_t)ks * SS + p) * TPAD + t] = c[s][r];
    }
  }
  #undef ISSUE_ASYNC_B
}

// ---------------------------------------------------------------------------
// Kernel 4: reduce k-slices, drop t padding, emit [48,48,550] f32
// ---------------------------------------------------------------------------
__global__ void __launch_bounds__(256)
reduce_kernel(const float* __restrict__ part, float* __restrict__ out) {
  const int idx = blockIdx.x * 256 + threadIdx.x;   // p*TT + t
  const int p = idx / TT;
  const int t = idx - p * TT;
  float s = 0.f;
  #pragma unroll
  for (int ksl = 0; ksl < NSLICE; ++ksl)
    s += part[((size_t)ksl * SS + p) * TPAD + t];
  out[idx] = s;
}

// ---------------------------------------------------------------------------
extern "C" void kernel_launch(void* const* d_in, const int* in_sizes, int n_in,
                              void* d_out, int out_size, void* d_ws, size_t ws_size,
                              hipStream_t stream) {
  (void)in_sizes; (void)n_in; (void)out_size; (void)ws_size;
  const float* xin  = (const float*)d_in[0];
  const float* z    = (const float*)d_in[1];
  const float* mask = (const float*)d_in[2];
  const float* sloc = (const float*)d_in[3];
  const float* W1   = (const float*)d_in[4];
  const float* b1   = (const float*)d_in[5];
  const float* W2   = (const float*)d_in[6];
  const float* b2   = (const float*)d_in[7];
  const float* W3   = (const float*)d_in[8];
  const float* b3   = (const float*)d_in[9];
  const float* elsp = (const float*)d_in[10];
  const float* nbs  = (const float*)d_in[11];

  char* ws = (char*)d_ws;
  float*    resp = (float*)(ws + WS_RESP);
  _Float16* evT  = (_Float16*)(ws + WS_EVT);
  float*    part = (float*)(ws + WS_PART);

  mlp_resp_kernel<<<NE / 8, 256, 0, stream>>>(xin, mask, W1, b1, W2, b2, W3, b3, resp);
  ev_kernel<<<(TPAD * NE) / 256, 256, 0, stream>>>(z, nbs, evT);
  gemm_kernel<<<dim3(SS / 64, TPAD / 192, NSLICE), 256, 0, stream>>>(
      xin, sloc, elsp, resp, (const unsigned*)evT, part);
  reduce_kernel<<<(SS * TT) / 256, 256, 0, stream>>>(part, (float*)d_out);
}
